// TensorNet_71313636983496
// MI455X (gfx1250) — matrix operand order repaired and checked by reading, not compiled, not run on hardware
//
#include <hip/hip_runtime.h>

#define NN 4096
#define NE 65536
#define H  128
#define RBF 32

typedef __attribute__((ext_vector_type(16))) __bf16 v16bf;
typedef __attribute__((ext_vector_type(8)))  __bf16 v8bf;
typedef __attribute__((ext_vector_type(8)))  float  v8f;
typedef unsigned short u16;

__device__ __forceinline__ u16 f2bf(float f) {
  union { float f; unsigned u; } v; v.f = f;
  unsigned r = v.u + 0x7fffu + ((v.u >> 16) & 1u);
  return (u16)(r >> 16);
}

__device__ __forceinline__ v16bf mk16(v8bf lo, v8bf hi) {
  return __builtin_shufflevector(lo, hi, 0,1,2,3,4,5,6,7,8,9,10,11,12,13,14,15);
}

__device__ __forceinline__ v8f wmma_bf16(v16bf a, v16bf b, v8f c) {
  return __builtin_amdgcn_wmma_f32_16x16x32_bf16(false, a, false, b, (short)0, c, false, false);
}

// Direct global -> LDS async copy of 16 bytes (CDNA5 TDM-adjacent path, ASYNCcnt).
// lds must point into __shared__; flat LDS address carries the LDS offset in [31:0].
__device__ __forceinline__ void async_ld16(void* lds, const void* gaddr) {
  unsigned loff = (unsigned)(unsigned long long)lds;
  asm volatile("global_load_async_to_lds_b128 %0, %1, off"
               :: "v"(loff), "v"(gaddr) : "memory");
}
__device__ __forceinline__ void async_wait0() {
  asm volatile("s_wait_asynccnt 0" ::: "memory");
}

// A fragment from a row-major bf16 panel: M=16 rows, ld elems/row.
// lane<16 (half=0): row=lane, K={k0..k0+15}
// lane>=16 (half=1): row=lane-16, K={k0+16..k0+31}
__device__ __forceinline__ v16bf ldA(const u16* panel, int ld, int k0, int mr, int half) {
  const u16* p = panel + mr * ld + k0 + half * 16;
  v8bf lo = *(const v8bf*)p;
  v8bf hi = *(const v8bf*)(p + 8);
  return mk16(lo, hi);
}

// B fragment: weights row-major [out][ldk]; B[k][n] = W[n][k]
// lane holds column n=n0+mr, 16 contiguous K starting at k0+half*16.
__device__ __forceinline__ v16bf ldB(const u16* W, int ldk, int n0, int k0, int mr, int half) {
  return *(const v16bf*)(W + (n0 + mr) * ldk + k0 + half * 16);
}

// ---------------- prep: node features -> bf16 ----------------
__global__ void k_prep_nodes(const int* z, const float* labels, const int* mask,
                             const float* emb_w, const float* label_emb_w,
                             const float* label_emb_b, const float* mask_token,
                             u16* Zbf, u16* Lbf) {
  int idx = blockIdx.x * blockDim.x + threadIdx.x;
  if (idx >= NN * H) return;
  int n = idx >> 7, h = idx & (H - 1);
  Zbf[idx] = f2bf(emb_w[z[n] * H + h]);
  float lv = mask[n] ? mask_token[h] : fmaf(labels[n], label_emb_w[h], label_emb_b[h]);
  Lbf[idx] = f2bf(lv);
}

// ---------------- prep: weights -> bf16 ----------------
__global__ void k_prep_w(const float* emb2_w, const float* label_emb2_w,
                         const float* emb2_b, const float* label_emb2_b,
                         const float* dp1_w, const float* dp2_w, const float* dp3_w,
                         const float* ls1_w, const float* ls2_w,
                         const float* lt0_w, const float* lt1_w, const float* lt2_w,
                         u16* Wcat, u16* dpW, u16* ls1bf, u16* ls2bf,
                         u16* lt0bf, u16* lt1bf, u16* lt2bf, float* bcat) {
  int i = blockIdx.x * blockDim.x + threadIdx.x;
  if (i < 65536) {  // Wcat[128][512] = [emb2_w | label_emb2_w]
    int h = i >> 9, k = i & 511;
    float w = (k < 256) ? emb2_w[h * 256 + k] : label_emb2_w[h * 256 + (k - 256)];
    Wcat[i] = f2bf(w);
    return;
  }
  i -= 65536;
  if (i < 12288) {  // dpW[384][32] = dp1;dp2;dp3
    int row = i >> 5, k = i & 31;
    float w = (row < 128) ? dp1_w[row * 32 + k]
            : (row < 256) ? dp2_w[(row - 128) * 32 + k]
                          : dp3_w[(row - 256) * 32 + k];
    dpW[i] = f2bf(w);
    return;
  }
  i -= 12288;
  if (i < 32768) { ls1bf[i] = f2bf(ls1_w[i]); return; }
  i -= 32768;
  if (i < 98304) { ls2bf[i] = f2bf(ls2_w[i]); return; }
  i -= 98304;
  if (i < 16384) { lt0bf[i] = f2bf(lt0_w[i]); return; }
  i -= 16384;
  if (i < 16384) { lt1bf[i] = f2bf(lt1_w[i]); return; }
  i -= 16384;
  if (i < 16384) { lt2bf[i] = f2bf(lt2_w[i]); return; }
  i -= 16384;
  if (i < 128) { bcat[i] = emb2_b[i] + label_emb2_b[i]; return; }
}

// ---------------- edge phase: fused message GEMM + structured scatter ----------------
__global__ __launch_bounds__(32)
void k_edges(const int* edge_index, const float* edge_weight, const float* edge_vec,
             const float* edge_attr,
             const u16* Zbf, const u16* Lbf, const u16* Wcat, const u16* dpW,
             const float* bcat, const float* dp1_b, const float* dp2_b, const float* dp3_b,
             float* accI, float* accA, float* accS) {
  __shared__ u16 ash[16 * 512];     // [Z[src]|Z[dst]|L[src]|L[dst]] per edge row
  __shared__ u16 eash[16 * 32];     // edge_attr bf16
  __shared__ float geom[16 * 10];   // cw, -v2, v1, -v0, s00,s11,s22,s01,s02,s12
  __shared__ int srcsh[16], dstsh[16];
  __shared__ float biasC[128];
  __shared__ float biasF[384];

  const int tid = threadIdx.x;
  const int e0 = blockIdx.x * 16;

  for (int i = tid; i < 128; i += 32) biasC[i] = bcat[i];
  for (int i = tid; i < 384; i += 32)
    biasF[i] = (i < 128) ? dp1_b[i] : (i < 256) ? dp2_b[i - 128] : dp3_b[i - 256];

  if (tid < 16) {
    int e = e0 + tid;
    srcsh[tid] = edge_index[e];
    dstsh[tid] = edge_index[NE + e];
    float w = edge_weight[e];
    float cw = (w < 4.5f) ? 0.5f * (__cosf(w * 0.6981317007977318f) + 1.0f) : 0.0f;
    float v0 = edge_vec[3 * e], v1 = edge_vec[3 * e + 1], v2 = edge_vec[3 * e + 2];
    float tr3 = (v0 * v0 + v1 * v1 + v2 * v2) * (1.0f / 3.0f);
    float* g = &geom[tid * 10];
    g[0] = cw; g[1] = -v2; g[2] = v1; g[3] = -v0;
    g[4] = v0 * v0 - tr3; g[5] = v1 * v1 - tr3; g[6] = v2 * v2 - tr3;
    g[7] = v0 * v1; g[8] = v0 * v2; g[9] = v1 * v2;
  }
  __syncthreads();

  // gather A panel: 16 rows x 512 bf16, streamed directly global -> LDS (ASYNCcnt)
  for (int it = tid; it < 1024; it += 32) {
    int row = it >> 6, rem = it & 63;
    int seg = rem >> 4, c = rem & 15;
    int node = (seg & 1) ? dstsh[row] : srcsh[row];
    const u16* srcarr = (seg < 2) ? Zbf : Lbf;
    async_ld16(&ash[row * 512 + seg * 128 + c * 8], &srcarr[node * H + c * 8]);
  }
  for (int it = tid; it < 512; it += 32)
    eash[it] = f2bf(edge_attr[e0 * RBF + it]);
  async_wait0();
  __syncthreads();

  const int half = tid >> 4, mr = tid & 15;
  #pragma unroll 1
  for (int ct = 0; ct < 8; ++ct) {
    v8f cz = {};
    #pragma unroll
    for (int kt = 0; kt < 16; ++kt) {
      v16bf a = ldA(ash, 512, kt * 32, mr, half);
      v16bf b = ldB(Wcat, 512, ct * 16, kt * 32, mr, half);
      cz = wmma_bf16(a, b, cz);
    }
    v8f f1a = {}, f2a = {}, f3a = {};
    v16bf ae = ldA(eash, 32, 0, mr, half);
    f1a = wmma_bf16(ae, ldB(dpW, 32, ct * 16, 0, mr, half), f1a);
    f2a = wmma_bf16(ae, ldB(dpW, 32, 128 + ct * 16, 0, mr, half), f2a);
    f3a = wmma_bf16(ae, ldB(dpW, 32, 256 + ct * 16, 0, mr, half), f3a);

    int ch = ct * 16 + mr;
    float bC = biasC[ch], b1v = biasF[ch], b2v = biasF[128 + ch], b3v = biasF[256 + ch];
    #pragma unroll
    for (int r = 0; r < 8; ++r) {
      int er = r + half * 8;
      const float* g = &geom[er * 10];
      float czv = g[0] * (cz[r] + bC);
      float Ic = (f1a[r] + b1v) * czv;
      float Ac = (f2a[r] + b2v) * czv;
      float Sc = (f3a[r] + b3v) * czv;
      int base = srcsh[er] * H + ch;
      unsafeAtomicAdd(&accI[base], Ic);
      unsafeAtomicAdd(&accA[base * 3 + 0], Ac * g[1]);
      unsafeAtomicAdd(&accA[base * 3 + 1], Ac * g[2]);
      unsafeAtomicAdd(&accA[base * 3 + 2], Ac * g[3]);
      #pragma unroll
      for (int c = 0; c < 6; ++c)
        unsafeAtomicAdd(&accS[base * 6 + c], Sc * g[4 + c]);
    }
  }
}

// ---------------- node norm + layernorm ----------------
__global__ __launch_bounds__(128)
void k_norm(const float* accI, const float* accA, const float* accS,
            const float* ln_g, const float* ln_b,
            u16* Ibf, u16* Abf, u16* Sbf, u16* normbf) {
  __shared__ float r1[128], r2[128];
  int n = blockIdx.x, h = threadIdx.x;
  int idx = n * H + h;
  float Ic = accI[idx];
  float a0 = accA[idx * 3], a1 = accA[idx * 3 + 1], a2 = accA[idx * 3 + 2];
  float s0 = accS[idx * 6],     s1 = accS[idx * 6 + 1], s2 = accS[idx * 6 + 2];
  float s3 = accS[idx * 6 + 3], s4 = accS[idx * 6 + 4], s5 = accS[idx * 6 + 5];
  Ibf[idx] = f2bf(Ic);
  Abf[0 * NN * H + idx] = f2bf(a0);
  Abf[1 * NN * H + idx] = f2bf(a1);
  Abf[2 * NN * H + idx] = f2bf(a2);
  Sbf[0 * NN * H + idx] = f2bf(s0);
  Sbf[1 * NN * H + idx] = f2bf(s1);
  Sbf[2 * NN * H + idx] = f2bf(s2);
  Sbf[3 * NN * H + idx] = f2bf(s3);
  Sbf[4 * NN * H + idx] = f2bf(s4);
  Sbf[5 * NN * H + idx] = f2bf(s5);
  // ||I+A+S||_F^2 in closed form
  float x = (Ic + s0) * (Ic + s0) + (Ic + s1) * (Ic + s1) + (Ic + s2) * (Ic + s2)
          + 2.0f * (a0 * a0 + a1 * a1 + a2 * a2 + s3 * s3 + s4 * s4 + s5 * s5);
  r1[h] = x; r2[h] = x * x;
  __syncthreads();
  for (int st = 64; st > 0; st >>= 1) {
    if (h < st) { r1[h] += r1[h + st]; r2[h] += r2[h + st]; }
    __syncthreads();
  }
  float mu  = r1[0] * (1.0f / 128.0f);
  float var = r2[0] * (1.0f / 128.0f) - mu * mu;
  float y = (x - mu) * rsqrtf(var + 1e-5f) * ln_g[h] + ln_b[h];
  normbf[idx] = f2bf(y);
}

// ---------------- gate MLP: silu(ls1) -> silu(ls2) ----------------
__global__ __launch_bounds__(32)
void k_gate(const u16* normbf, const u16* ls1bf, const u16* ls2bf,
            const float* ls1_b, const float* ls2_b, float* gate) {
  __shared__ u16 nsh[16 * 128];
  __shared__ u16 g1sh[16 * 256];
  int tid = threadIdx.x, n0 = blockIdx.x * 16;
  for (int it = tid; it < 256; it += 32)
    async_ld16(&nsh[it * 8], &normbf[n0 * H + it * 8]);
  async_wait0();
  __syncthreads();
  int half = tid >> 4, mr = tid & 15;
  #pragma unroll 1
  for (int ct = 0; ct < 16; ++ct) {
    v8f acc = {};
    #pragma unroll
    for (int kt = 0; kt < 4; ++kt)
      acc = wmma_bf16(ldA(nsh, 128, kt * 32, mr, half),
                      ldB(ls1bf, 128, ct * 16, kt * 32, mr, half), acc);
    int ch = ct * 16 + mr;
    float bb = ls1_b[ch];
    #pragma unroll
    for (int r = 0; r < 8; ++r) {
      float xv = acc[r] + bb;
      float sv = xv / (1.0f + __expf(-xv));
      g1sh[(r + half * 8) * 256 + ch] = f2bf(sv);
    }
  }
  __syncthreads();
  #pragma unroll 1
  for (int ct = 0; ct < 24; ++ct) {
    v8f acc = {};
    #pragma unroll
    for (int kt = 0; kt < 8; ++kt)
      acc = wmma_bf16(ldA(g1sh, 256, kt * 32, mr, half),
                      ldB(ls2bf, 256, ct * 16, kt * 32, mr, half), acc);
    int ch = ct * 16 + mr;
    float bb = ls2_b[ch];
    #pragma unroll
    for (int r = 0; r < 8; ++r) {
      float xv = acc[r] + bb;
      gate[(n0 + r + half * 8) * 384 + ch] = xv / (1.0f + __expf(-xv));
    }
  }
}

// ---------------- channel mixing (10 GEMMs) + gated 3x3 composition ----------------
__global__ __launch_bounds__(32)
void k_mix(const u16* Ibf, const u16* Abf, const u16* Sbf,
           const u16* lt0bf, const u16* lt1bf, const u16* lt2bf,
           const float* gate, float* out) {
  __shared__ u16 psh[10 * 16 * 128];  // 10 component panels, 16 nodes x 128 ch
  int tid = threadIdx.x, n0 = blockIdx.x * 16;
  for (int it = tid; it < 2560; it += 32) {
    int p = it >> 8, within = it & 255;
    const u16* src = (p == 0) ? (Ibf + n0 * H)
                   : (p < 4)  ? (Abf + (p - 1) * (NN * H) + n0 * H)
                              : (Sbf + (p - 4) * (NN * H) + n0 * H);
    async_ld16(&psh[p * 2048 + within * 8], &src[within * 8]);
  }
  async_wait0();
  __syncthreads();
  int half = tid >> 4, mr = tid & 15;
  #pragma unroll 1
  for (int ct = 0; ct < 8; ++ct) {
    v8f acc[10] = {};
    #pragma unroll
    for (int kt = 0; kt < 4; ++kt) {
      int k0 = kt * 32;
      v16bf bI = ldB(lt0bf, 128, ct * 16, k0, mr, half);
      v16bf bA = ldB(lt1bf, 128, ct * 16, k0, mr, half);
      v16bf bS = ldB(lt2bf, 128, ct * 16, k0, mr, half);
      acc[0] = wmma_bf16(ldA(psh, 128, k0, mr, half), bI, acc[0]);
      #pragma unroll
      for (int c = 1; c < 4; ++c)
        acc[c] = wmma_bf16(ldA(psh + c * 2048, 128, k0, mr, half), bA, acc[c]);
      #pragma unroll
      for (int c = 4; c < 10; ++c)
        acc[c] = wmma_bf16(ldA(psh + c * 2048, 128, k0, mr, half), bS, acc[c]);
    }
    int g = ct * 16 + mr;
    #pragma unroll
    for (int r = 0; r < 8; ++r) {
      int node = n0 + r + half * 8;
      const float* gp = &gate[node * 384 + g * 3];
      float g0 = gp[0], g1 = gp[1], g2 = gp[2];
      float Im  = acc[0][r] * g0;
      float a01 = acc[1][r] * g1, a02 = acc[2][r] * g1, a12 = acc[3][r] * g1;
      float s00 = acc[4][r] * g2, s11 = acc[5][r] * g2, s22 = acc[6][r] * g2;
      float s01 = acc[7][r] * g2, s02 = acc[8][r] * g2, s12 = acc[9][r] * g2;
      float* o = out + (node * H + g) * 9;
      o[0] = Im + s00;  o[1] = a01 + s01; o[2] = a02 + s02;
      o[3] = s01 - a01; o[4] = Im + s11; o[5] = a12 + s12;
      o[6] = s02 - a02; o[7] = s12 - a12; o[8] = Im + s22;
    }
  }
}

extern "C" void kernel_launch(void* const* d_in, const int* in_sizes, int n_in,
                              void* d_out, int out_size, void* d_ws, size_t ws_size,
                              hipStream_t stream) {
  const int*   z          = (const int*)  d_in[0];
  const int*   edge_index = (const int*)  d_in[1];
  const float* edge_w     = (const float*)d_in[2];
  const float* edge_vec   = (const float*)d_in[3];
  const float* edge_attr  = (const float*)d_in[4];
  const float* labels     = (const float*)d_in[5];
  const int*   lmask      = (const int*)  d_in[6];
  const float* emb_w      = (const float*)d_in[7];
  const float* emb2_w     = (const float*)d_in[8];
  const float* emb2_b     = (const float*)d_in[9];
  const float* lemb_w     = (const float*)d_in[10];
  const float* lemb_b     = (const float*)d_in[11];
  const float* mask_tok   = (const float*)d_in[12];
  const float* lemb2_w    = (const float*)d_in[13];
  const float* lemb2_b    = (const float*)d_in[14];
  const float* dp1_w      = (const float*)d_in[15];
  const float* dp1_b      = (const float*)d_in[16];
  const float* dp2_w      = (const float*)d_in[17];
  const float* dp2_b      = (const float*)d_in[18];
  const float* dp3_w      = (const float*)d_in[19];
  const float* dp3_b      = (const float*)d_in[20];
  const float* lt0_w      = (const float*)d_in[21];
  const float* lt1_w      = (const float*)d_in[22];
  const float* lt2_w      = (const float*)d_in[23];
  const float* ls1_w      = (const float*)d_in[24];
  const float* ls1_b      = (const float*)d_in[25];
  const float* ls2_w      = (const float*)d_in[26];
  const float* ls2_b      = (const float*)d_in[27];
  const float* ln_g       = (const float*)d_in[28];
  const float* ln_b       = (const float*)d_in[29];

  char* w = (char*)d_ws;
  size_t o = 0;
  float* accI = (float*)(w + o); o += (size_t)NN * H * 4;
  float* accA = (float*)(w + o); o += (size_t)NN * H * 3 * 4;
  float* accS = (float*)(w + o); o += (size_t)NN * H * 6 * 4;
  size_t zero_bytes = o;
  u16* Zbf    = (u16*)(w + o); o += (size_t)NN * H * 2;
  u16* Lbf    = (u16*)(w + o); o += (size_t)NN * H * 2;
  u16* normbf = (u16*)(w + o); o += (size_t)NN * H * 2;
  u16* Ibf    = (u16*)(w + o); o += (size_t)NN * H * 2;
  u16* Abf    = (u16*)(w + o); o += (size_t)NN * H * 3 * 2;
  u16* Sbf    = (u16*)(w + o); o += (size_t)NN * H * 6 * 2;
  u16* Wcat   = (u16*)(w + o); o += (size_t)128 * 512 * 2;
  u16* dpW    = (u16*)(w + o); o += (size_t)384 * 32 * 2;
  u16* ls1bf  = (u16*)(w + o); o += (size_t)256 * 128 * 2;
  u16* ls2bf  = (u16*)(w + o); o += (size_t)384 * 256 * 2;
  u16* lt0bf  = (u16*)(w + o); o += (size_t)128 * 128 * 2;
  u16* lt1bf  = (u16*)(w + o); o += (size_t)128 * 128 * 2;
  u16* lt2bf  = (u16*)(w + o); o += (size_t)128 * 128 * 2;
  float* bcat = (float*)(w + o); o += (size_t)128 * 4;
  float* gate = (float*)(w + o); o += (size_t)NN * 384 * 4;

  hipMemsetAsync(d_ws, 0, zero_bytes, stream);

  k_prep_nodes<<<(NN * H + 255) / 256, 256, 0, stream>>>(
      z, labels, lmask, emb_w, lemb_w, lemb_b, mask_tok, Zbf, Lbf);

  k_prep_w<<<(258176 + 255) / 256, 256, 0, stream>>>(
      emb2_w, lemb2_w, emb2_b, lemb2_b, dp1_w, dp2_w, dp3_w,
      ls1_w, ls2_w, lt0_w, lt1_w, lt2_w,
      Wcat, dpW, ls1bf, ls2bf, lt0bf, lt1bf, lt2bf, bcat);

  k_edges<<<NE / 16, 32, 0, stream>>>(
      edge_index, edge_w, edge_vec, edge_attr,
      Zbf, Lbf, Wcat, dpW, bcat, dp1_b, dp2_b, dp3_b,
      accI, accA, accS);

  k_norm<<<NN, 128, 0, stream>>>(accI, accA, accS, ln_g, ln_b,
                                 Ibf, Abf, Sbf, normbf);

  k_gate<<<NN / 16, 32, 0, stream>>>(normbf, ls1bf, ls2bf, ls1_b, ls2_b, gate);

  k_mix<<<NN / 16, 32, 0, stream>>>(Ibf, Abf, Sbf, lt0bf, lt1bf, lt2bf,
                                    gate, (float*)d_out);
}
